// MHA_72413148610860
// MI455X (gfx1250) — matrix operand order repaired and checked by reading, not compiled, not run on hardware
//
#include <hip/hip_runtime.h>

typedef _Float16 half_t;
typedef __attribute__((ext_vector_type(16))) _Float16 v16h;
typedef __attribute__((ext_vector_type(8)))  _Float16 v8h;
typedef __attribute__((ext_vector_type(8)))  float    v8f;

#define BDIM   2
#define SEQ    2048
#define HDIM   512
#define NH     8
#define DH     64
#define BS     (BDIM * SEQ)     // 4096
#define ND     (NH * DH)        // 512
#define CHUNK  64
#define NCHUNK (SEQ / CHUNK)    // 32
#define BN     (BDIM * NH)      // 16
#define EPSV   1e-6f

#define LDA 40   // padded stride (halves) for 32-wide K tiles: 80B rows, 16B aligned
#define LDK 72   // padded stride (halves) for 64-wide tiles: 144B rows, 16B aligned
#define LDF 68   // padded stride (floats) for f32 spill tiles: 272B rows, 16B aligned

#define USE_ASYNC_LDS 1

static __device__ inline int lane_id() { return (int)(threadIdx.x & 31); }

// ---------------- CDNA5 wave32 WMMA fragment loaders (vectorized) ----------------
// A: 16x32 (MxK) f16, row-major [M][K]. Lane l: m=l&15, kb=(l>>4)*8;
// elements 0..7 = K kb..kb+7, 8..15 = K 16+kb..16+kb+7  -> 2x ds_load_b128.
static __device__ inline v16h frag_a(const half_t* tile, int row0, int lda, int k0) {
  const int l = lane_id();
  const half_t* p = tile + (size_t)(row0 + (l & 15)) * lda + k0 + ((l >> 4) * 8);
  v8h lo = *(const v8h*)p;
  v8h hi = *(const v8h*)(p + 16);
  return __builtin_shufflevector(lo, hi, 0,1,2,3,4,5,6,7,8,9,10,11,12,13,14,15);
}

// B: 32x16 (KxN) f16, stored transposed as bt[n][k]. Lane l: n=l&15, kb=(l>>4)*8;
// elements 0..7 = K kb..kb+7, 8..15 = K 16+kb..16+kb+7 -> 2x ds_load_b128.
static __device__ inline v16h frag_bt(const half_t* bt, int col0, int ldbt, int k0) {
  const int l = lane_id();
  const half_t* p = bt + (size_t)(col0 + (l & 15)) * ldbt + k0 + ((l >> 4) * 8);
  v8h lo = *(const v8h*)p;
  v8h hi = *(const v8h*)(p + 16);
  return __builtin_shufflevector(lo, hi, 0,1,2,3,4,5,6,7,8,9,10,11,12,13,14,15);
}

#define WMMA(a, b, c) __builtin_amdgcn_wmma_f32_16x16x32_f16(false, (a), false, (b), (short)0, (c), false, false)

static __device__ inline void cvt_store8(half_t* dst, float4 a, float4 b) {
  v8h h;
  h[0] = (half_t)a.x; h[1] = (half_t)a.y; h[2] = (half_t)a.z; h[3] = (half_t)a.w;
  h[4] = (half_t)b.x; h[5] = (half_t)b.y; h[6] = (half_t)b.z; h[7] = (half_t)b.w;
  *(v8h*)dst = h;
}

static __device__ inline float feature_map(float x) {
  return (x > 0.0f) ? (x + 1.0f) : __expf(x);
}

// ---- async global->LDS (ASYNCcnt path); LDS offset = low 32 bits of generic addr ----
static __device__ inline void async_copy16(half_t* lds_dst, const half_t* gsrc) {
#if USE_ASYNC_LDS
  unsigned ldso = (unsigned)(unsigned long long)(size_t)lds_dst;
  unsigned long long ga = (unsigned long long)(size_t)gsrc;
  asm volatile("global_load_async_to_lds_b128 %0, %1, off"
               :: "v"(ldso), "v"(ga) : "memory");
#else
  *(v8h*)lds_dst = *(const v8h*)gsrc;
#endif
}
static __device__ inline void async_join() {
#if USE_ASYNC_LDS
  asm volatile("s_wait_asynccnt 0" ::: "memory");
#endif
}

// =====================================================================
// K0: transpose+convert 512x512 weights: out_f16[n][k] = in_f32[k][n]
// grid = (8, 8, 4)
// =====================================================================
__global__ __launch_bounds__(256) void transpose_w_kernel(
    const float* __restrict__ wq, const float* __restrict__ wk,
    const float* __restrict__ wv, const float* __restrict__ wo,
    half_t* __restrict__ wTq, half_t* __restrict__ wTk,
    half_t* __restrict__ wTv, half_t* __restrict__ wTo) {
  __shared__ float tile[64 * 65];
  const int which = blockIdx.z;
  const float* in = (which == 0) ? wq : (which == 1) ? wk : (which == 2) ? wv : wo;
  half_t* out     = (which == 0) ? wTq : (which == 1) ? wTk : (which == 2) ? wTv : wTo;
  const int r0 = blockIdx.x * 64;
  const int c0 = blockIdx.y * 64;
  for (int j = threadIdx.x; j < 64 * 16; j += 256) {
    int r = j >> 4, c4 = (j & 15) * 4;
    float4 f = *(const float4*)(in + (size_t)(r0 + r) * 512 + c0 + c4);
    tile[r * 65 + c4 + 0] = f.x; tile[r * 65 + c4 + 1] = f.y;
    tile[r * 65 + c4 + 2] = f.z; tile[r * 65 + c4 + 3] = f.w;
  }
  __syncthreads();
  for (int j = threadIdx.x; j < 64 * 8; j += 256) {
    int rr = j >> 3, c8 = (j & 7) * 8;
    v8h h;
#pragma unroll
    for (int u = 0; u < 8; ++u) h[u] = (half_t)tile[(c8 + u) * 65 + rr];
    *(v8h*)(out + (size_t)(c0 + rr) * 512 + r0 + c8) = h;
  }
}

// =====================================================================
// K1: QKV projection, 128x64 block tile, 2x2 tiles/wave, double-buffered
// grid = (BS/128, ND/64, 3); out layout [B,N,S,D] f16
// =====================================================================
__global__ __launch_bounds__(256) void qkv_proj_kernel(
    const float* __restrict__ xq, const float* __restrict__ xk, const float* __restrict__ xv,
    const half_t* __restrict__ wTq, const half_t* __restrict__ wTk, const half_t* __restrict__ wTv,
    const float* __restrict__ bq, const float* __restrict__ bk, const float* __restrict__ bv,
    half_t* __restrict__ phi_q, half_t* __restrict__ phi_k, half_t* __restrict__ v_h) {
  __shared__ half_t ldsA[2][128 * LDA];
  __shared__ half_t ldsB[2][64 * LDA];
  __shared__ half_t ldsOut[128 * LDK];
  const int m0 = blockIdx.x * 128;
  const int n0 = blockIdx.y * 64;
  const int which = blockIdx.z;
  const float*  x    = (which == 0) ? xq  : (which == 1) ? xk  : xv;
  const half_t* wT   = (which == 0) ? wTq : (which == 1) ? wTk : wTv;
  const float*  bias = (which == 0) ? bq  : (which == 1) ? bk  : bv;
  half_t* out        = (which == 0) ? phi_q : (which == 1) ? phi_k : v_h;

  const int tid = threadIdx.x;
  const int wave = tid >> 5;
  const int tm0 = (wave >> 1) * 2, tn0 = (wave & 1) * 2;
  const int ar = tid >> 2, ac = (tid & 3) * 8;   // 64 rows x 4 chunks; A covers ar and ar+64
  v8f acc[2][2] = {};

  float4 fa0, fa1, fb0, fb1;
  v8h wreg;
  {
    const float4* xp0 = (const float4*)(x + (size_t)(m0 + ar) * HDIM + ac);
    fa0 = xp0[0]; fa1 = xp0[1];
    const float4* xp1 = (const float4*)(x + (size_t)(m0 + 64 + ar) * HDIM + ac);
    fb0 = xp1[0]; fb1 = xp1[1];
    wreg = *(const v8h*)&wT[(size_t)(n0 + ar) * HDIM + ac];
  }
  int p = 0;
  for (int k0 = 0; k0 < HDIM; k0 += 32) {
    cvt_store8(&ldsA[p][ar * LDA + ac], fa0, fa1);
    cvt_store8(&ldsA[p][(ar + 64) * LDA + ac], fb0, fb1);
    *(v8h*)&ldsB[p][ar * LDA + ac] = wreg;
    __syncthreads();
    const int kn = k0 + 32;
    if (kn < HDIM) {   // prefetch next stage while WMMAs run
      const float4* xp0 = (const float4*)(x + (size_t)(m0 + ar) * HDIM + kn + ac);
      fa0 = xp0[0]; fa1 = xp0[1];
      const float4* xp1 = (const float4*)(x + (size_t)(m0 + 64 + ar) * HDIM + kn + ac);
      fb0 = xp1[0]; fb1 = xp1[1];
      wreg = *(const v8h*)&wT[(size_t)(n0 + ar) * HDIM + kn + ac];
    }
    v16h a0 = frag_a(&ldsA[p][0], (tm0 + 0) * 16, LDA, 0);
    v16h a1 = frag_a(&ldsA[p][0], (tm0 + 1) * 16, LDA, 0);
    v16h b0 = frag_bt(&ldsB[p][0], (tn0 + 0) * 16, LDA, 0);
    v16h b1 = frag_bt(&ldsB[p][0], (tn0 + 1) * 16, LDA, 0);
    acc[0][0] = WMMA(a0, b0, acc[0][0]);
    acc[0][1] = WMMA(a0, b1, acc[0][1]);
    acc[1][0] = WMMA(a1, b0, acc[1][0]);
    acc[1][1] = WMMA(a1, b1, acc[1][1]);
    p ^= 1;
  }
  __syncthreads();

  // epilogue: bias + feature map, spill f16 to LDS, coalesced b128 stores
  const int l = lane_id();
  const int nn = l & 15;
  const int mb = (l >> 4) * 8;
#pragma unroll
  for (int i = 0; i < 2; ++i)
#pragma unroll
    for (int j = 0; j < 2; ++j)
#pragma unroll
      for (int jj = 0; jj < 8; ++jj) {
        int rm  = (tm0 + i) * 16 + mb + jj;
        int col = (tn0 + j) * 16 + nn;
        float val = acc[i][j][jj] + bias[n0 + col];
        if (which < 2) val = feature_map(val);
        ldsOut[rm * LDK + col] = (half_t)val;
      }
  __syncthreads();
  const int nh = n0 >> 6;   // all 64 cols of this block map to one head
  for (int j2 = tid; j2 < 128 * 8; j2 += 256) {
    int r = j2 >> 3, c8 = (j2 & 7) * 8;
    int m = m0 + r, b = m >> 11, s = m & (SEQ - 1);
    *(v8h*)&out[(((size_t)(b * NH + nh) * SEQ) + s) * DH + c8] =
        *(const v8h*)&ldsOut[r * LDK + c8];
  }
}

// =====================================================================
// K2: per-chunk KV = phiK^T * V (64x64) and z_c. grid = (BN, NCHUNK)
// =====================================================================
__global__ __launch_bounds__(256) void chunk_kv_kernel(
    const half_t* __restrict__ phi_k, const half_t* __restrict__ v_h,
    float* __restrict__ kv, float* __restrict__ zc) {
  __shared__ half_t ldsKT[DH * LDK];   // [d][s]   A operand
  __shared__ half_t ldsVT[DH * LDK];   // [dv][s]  BT operand
  const int bn = blockIdx.x;
  const int c  = blockIdx.y;
  const size_t base = ((size_t)bn * SEQ + (size_t)c * CHUNK) * DH;

  for (int i = threadIdx.x; i < CHUNK * 8; i += 256) {
    int s = i >> 3, d8 = (i & 7) * 8;
    v8h kk = *(const v8h*)&phi_k[base + (size_t)s * DH + d8];
    v8h vv = *(const v8h*)&v_h[base + (size_t)s * DH + d8];
#pragma unroll
    for (int u = 0; u < 8; ++u) {
      ldsKT[(d8 + u) * LDK + s] = kk[u];
      ldsVT[(d8 + u) * LDK + s] = vv[u];
    }
  }
  __syncthreads();

  const int wave = threadIdx.x >> 5;
  const int tm = wave >> 1, tn0 = (wave & 1) * 2;
  v8f acc[2] = {{}, {}};
#pragma unroll
  for (int k0 = 0; k0 < CHUNK; k0 += 32) {
    v16h a = frag_a(ldsKT, tm * 16, LDK, k0);
#pragma unroll
    for (int tt = 0; tt < 2; ++tt) {
      v16h b = frag_bt(ldsVT, (tn0 + tt) * 16, LDK, k0);
      acc[tt] = WMMA(a, b, acc[tt]);
    }
  }
  const int l = lane_id();
  const int nn = l & 15;
  const int mb = (l >> 4) * 8;
  float* kvp = kv + ((size_t)bn * NCHUNK + c) * DH * DH;
#pragma unroll
  for (int tt = 0; tt < 2; ++tt)
#pragma unroll
    for (int j = 0; j < 8; ++j)
      kvp[(size_t)(tm * 16 + mb + j) * DH + (tn0 + tt) * 16 + nn] = acc[tt][j];
  if (threadIdx.x < DH) {
    int d = threadIdx.x;
    float sum = 0.f;
    for (int s = 0; s < CHUNK; ++s) sum += (float)ldsKT[d * LDK + s];
    zc[((size_t)bn * NCHUNK + c) * DH + d] = sum;
  }
}

// =====================================================================
// K3: exclusive prefix scan of chunk states per (b,n). grid = BN
// =====================================================================
__global__ __launch_bounds__(256) void scan_kernel(
    const float* __restrict__ kv, const float* __restrict__ zc,
    half_t* __restrict__ stateH, float* __restrict__ zx) {
  const int bn = blockIdx.x;
  for (int e = threadIdx.x; e < DH * DH; e += 256) {
    float run = 0.f;
    for (int c = 0; c < NCHUNK; ++c) {
      size_t idx = ((size_t)bn * NCHUNK + c) * DH * DH + e;
      stateH[idx] = (half_t)run;   // exclusive
      run += kv[idx];
    }
  }
  if (threadIdx.x < DH) {
    int d = threadIdx.x;
    float run = 0.f;
    for (int c = 0; c < NCHUNK; ++c) {
      size_t idx = ((size_t)bn * NCHUNK + c) * DH + d;
      zx[idx] = run;               // exclusive
      run += zc[idx];
    }
  }
}

// =====================================================================
// K4: per-chunk output. A = causal(phiQ phiK^T); O = A*V + phiQ*State;
// qz[s] = rowsum(A) + phiQ[s].z_excl + eps.  grid = (BN, NCHUNK)
// =====================================================================
__global__ __launch_bounds__(256) void attn_kernel(
    const half_t* __restrict__ phi_q, const half_t* __restrict__ phi_k,
    const half_t* __restrict__ v_h, const half_t* __restrict__ stateH,
    const float* __restrict__ zx, half_t* __restrict__ av) {
  __shared__ half_t ldsQ[CHUNK * LDK];    // [s][d]   A operand
  __shared__ half_t ldsK[CHUNK * LDK];    // [s'][d]  = BT for QK^T
  __shared__ half_t ldsVT[DH * LDK];      // [dv][s'] = BT for A*V
  __shared__ half_t ldsStT[DH * LDK];     // [dout][d]= BT for Q*State
  __shared__ half_t ldsAh[CHUNK * LDK];   // [s][s']  A operand
  __shared__ float  ldsAf[CHUNK * LDK];
  __shared__ float  ldsQZ[CHUNK];
  const int bn = blockIdx.x;
  const int c  = blockIdx.y;
  const int b  = bn >> 3;
  const int nh = bn & 7;
  const size_t base   = ((size_t)bn * SEQ + (size_t)c * CHUNK) * DH;
  const size_t stbase = ((size_t)bn * NCHUNK + c) * DH * DH;

  // Q, K rows: contiguous f16 -> async DMA to LDS
  for (int i = threadIdx.x; i < CHUNK * 8; i += 256) {
    int s = i >> 3, d8 = (i & 7) * 8;
    async_copy16(&ldsQ[s * LDK + d8], &phi_q[base + (size_t)s * DH + d8]);
    async_copy16(&ldsK[s * LDK + d8], &phi_k[base + (size_t)s * DH + d8]);
  }
  // V, State need transposes: vector global read + scattered LDS store
  for (int i = threadIdx.x; i < CHUNK * 8; i += 256) {
    int s = i >> 3, d8 = (i & 7) * 8;
    v8h vv = *(const v8h*)&v_h[base + (size_t)s * DH + d8];
    v8h st = *(const v8h*)&stateH[stbase + (size_t)s * DH + d8];
#pragma unroll
    for (int u = 0; u < 8; ++u) {
      ldsVT[(d8 + u) * LDK + s] = vv[u];
      ldsStT[(d8 + u) * LDK + s] = st[u];
    }
  }
  async_join();
  __syncthreads();

  const int wave = threadIdx.x >> 5;
  const int tm = wave >> 1, tn0 = (wave & 1) * 2;
  const int l = lane_id();
  const int nn = l & 15;
  const int mb = (l >> 4) * 8;

  // ---- A = phiQ * phiK^T, causal mask ----
  {
    v8f acc[2] = {{}, {}};
#pragma unroll
    for (int k0 = 0; k0 < DH; k0 += 32) {
      v16h a = frag_a(ldsQ, tm * 16, LDK, k0);
#pragma unroll
      for (int tt = 0; tt < 2; ++tt) {
        v16h bfr = frag_bt(ldsK, (tn0 + tt) * 16, LDK, k0);
        acc[tt] = WMMA(a, bfr, acc[tt]);
      }
    }
#pragma unroll
    for (int tt = 0; tt < 2; ++tt)
#pragma unroll
      for (int j = 0; j < 8; ++j) {
        int s  = tm * 16 + mb + j;
        int sp = (tn0 + tt) * 16 + nn;
        float vvv = (sp <= s) ? acc[tt][j] : 0.0f;  // causal
        ldsAf[s * LDK + sp] = vvv;
        ldsAh[s * LDK + sp] = (half_t)vvv;
      }
  }
  __syncthreads();

  if (threadIdx.x < CHUNK) {
    int s = threadIdx.x;
    float qz = EPSV;
    const float* zp = zx + ((size_t)bn * NCHUNK + c) * DH;
    for (int sp = 0; sp < CHUNK; ++sp) qz += ldsAf[s * LDK + sp];
    for (int d = 0; d < DH; ++d) qz += (float)ldsQ[s * LDK + d] * zp[d];
    ldsQZ[s] = qz;
  }
  __syncthreads();

  // ---- O = A_masked * V + phiQ * State_excl ----
  {
    v8f acc[2] = {{}, {}};
#pragma unroll
    for (int k0 = 0; k0 < CHUNK; k0 += 32) {
      v16h a = frag_a(ldsAh, tm * 16, LDK, k0);
#pragma unroll
      for (int tt = 0; tt < 2; ++tt) {
        v16h bfr = frag_bt(ldsVT, (tn0 + tt) * 16, LDK, k0);
        acc[tt] = WMMA(a, bfr, acc[tt]);
      }
    }
#pragma unroll
    for (int k0 = 0; k0 < DH; k0 += 32) {
      v16h a = frag_a(ldsQ, tm * 16, LDK, k0);
#pragma unroll
      for (int tt = 0; tt < 2; ++tt) {
        v16h bfr = frag_bt(ldsStT, (tn0 + tt) * 16, LDK, k0);
        acc[tt] = WMMA(a, bfr, acc[tt]);
      }
    }
#pragma unroll
    for (int tt = 0; tt < 2; ++tt)
#pragma unroll
      for (int j = 0; j < 8; ++j) {
        int s = tm * 16 + mb + j;
        int d = (tn0 + tt) * 16 + nn;
        float o = acc[tt][j] / ldsQZ[s];
        int sg = c * CHUNK + s;
        av[(((size_t)b * SEQ + sg) * NH + nh) * DH + d] = (half_t)o;  // [B,S,N*D]
      }
  }
}

// =====================================================================
// K5: output projection [BS,ND] f16 x [ND,H] + bias -> fp32, 128x64 tile,
// 2x2 tiles/wave, double-buffered. grid = (BS/128, HDIM/64)
// =====================================================================
__global__ __launch_bounds__(256) void out_proj_kernel(
    const half_t* __restrict__ av, const half_t* __restrict__ wTo,
    const float* __restrict__ bo, float* __restrict__ out) {
  __shared__ half_t ldsA[2][128 * LDA];
  __shared__ half_t ldsB[2][64 * LDA];
  __shared__ float  ldsOutF[128 * LDF];
  const int m0 = blockIdx.x * 128;
  const int n0 = blockIdx.y * 64;
  const int tid = threadIdx.x;
  const int wave = tid >> 5;
  const int tm0 = (wave >> 1) * 2, tn0 = (wave & 1) * 2;
  const int ar = tid >> 2, ac = (tid & 3) * 8;
  v8f acc[2][2] = {};

  v8h areg0, areg1, wreg;
  areg0 = *(const v8h*)&av[(size_t)(m0 + ar) * ND + ac];
  areg1 = *(const v8h*)&av[(size_t)(m0 + 64 + ar) * ND + ac];
  wreg  = *(const v8h*)&wTo[(size_t)(n0 + ar) * ND + ac];
  int p = 0;
  for (int k0 = 0; k0 < ND; k0 += 32) {
    *(v8h*)&ldsA[p][ar * LDA + ac]        = areg0;
    *(v8h*)&ldsA[p][(ar + 64) * LDA + ac] = areg1;
    *(v8h*)&ldsB[p][ar * LDA + ac]        = wreg;
    __syncthreads();
    const int kn = k0 + 32;
    if (kn < ND) {
      areg0 = *(const v8h*)&av[(size_t)(m0 + ar) * ND + kn + ac];
      areg1 = *(const v8h*)&av[(size_t)(m0 + 64 + ar) * ND + kn + ac];
      wreg  = *(const v8h*)&wTo[(size_t)(n0 + ar) * ND + kn + ac];
    }
    v16h a0 = frag_a(&ldsA[p][0], (tm0 + 0) * 16, LDA, 0);
    v16h a1 = frag_a(&ldsA[p][0], (tm0 + 1) * 16, LDA, 0);
    v16h b0 = frag_bt(&ldsB[p][0], (tn0 + 0) * 16, LDA, 0);
    v16h b1 = frag_bt(&ldsB[p][0], (tn0 + 1) * 16, LDA, 0);
    acc[0][0] = WMMA(a0, b0, acc[0][0]);
    acc[0][1] = WMMA(a0, b1, acc[0][1]);
    acc[1][0] = WMMA(a1, b0, acc[1][0]);
    acc[1][1] = WMMA(a1, b1, acc[1][1]);
    p ^= 1;
  }
  __syncthreads();

  const int l = lane_id();
  const int nn = l & 15;
  const int mb = (l >> 4) * 8;
#pragma unroll
  for (int i = 0; i < 2; ++i)
#pragma unroll
    for (int j = 0; j < 2; ++j)
#pragma unroll
      for (int jj = 0; jj < 8; ++jj)
        ldsOutF[((tm0 + i) * 16 + mb + jj) * LDF + (tn0 + j) * 16 + nn] = acc[i][j][jj];
  __syncthreads();
  for (int j2 = tid; j2 < 128 * 16; j2 += 256) {
    int r = j2 >> 4, c4 = (j2 & 15) * 4;
    float4 o;
    o.x = ldsOutF[r * LDF + c4 + 0] + bo[n0 + c4 + 0];
    o.y = ldsOutF[r * LDF + c4 + 1] + bo[n0 + c4 + 1];
    o.z = ldsOutF[r * LDF + c4 + 2] + bo[n0 + c4 + 2];
    o.w = ldsOutF[r * LDF + c4 + 3] + bo[n0 + c4 + 3];
    *(float4*)&out[(size_t)(m0 + r) * HDIM + n0 + c4] = o;
  }
}

extern "C" void kernel_launch(void* const* d_in, const int* in_sizes, int n_in,
                              void* d_out, int out_size, void* d_ws, size_t ws_size,
                              hipStream_t stream) {
  const float* query = (const float*)d_in[0];
  const float* key   = (const float*)d_in[1];
  const float* value = (const float*)d_in[2];
  const float* wq = (const float*)d_in[3];
  const float* bq = (const float*)d_in[4];
  const float* wk = (const float*)d_in[5];
  const float* bk = (const float*)d_in[6];
  const float* wv = (const float*)d_in[7];
  const float* bv = (const float*)d_in[8];
  const float* wo = (const float*)d_in[9];
  const float* bo = (const float*)d_in[10];
  float* out = (float*)d_out;

  char* ws = (char*)d_ws;
  size_t off = 0;
  half_t* phi_q = (half_t*)(ws + off); off += (size_t)BN * SEQ * DH * sizeof(half_t);
  half_t* phi_k = (half_t*)(ws + off); off += (size_t)BN * SEQ * DH * sizeof(half_t);
  half_t* v_h   = (half_t*)(ws + off); off += (size_t)BN * SEQ * DH * sizeof(half_t);
  half_t* av    = (half_t*)(ws + off); off += (size_t)BS * ND * sizeof(half_t);
  float*  kv    = (float*) (ws + off); off += (size_t)BN * NCHUNK * DH * DH * sizeof(float);
  half_t* stH   = (half_t*)(ws + off); off += (size_t)BN * NCHUNK * DH * DH * sizeof(half_t);
  float*  zc    = (float*) (ws + off); off += (size_t)BN * NCHUNK * DH * sizeof(float);
  float*  zx    = (float*) (ws + off); off += (size_t)BN * NCHUNK * DH * sizeof(float);
  half_t* wTq   = (half_t*)(ws + off); off += (size_t)HDIM * ND * sizeof(half_t);
  half_t* wTk   = (half_t*)(ws + off); off += (size_t)HDIM * ND * sizeof(half_t);
  half_t* wTv   = (half_t*)(ws + off); off += (size_t)HDIM * ND * sizeof(half_t);
  half_t* wTo   = (half_t*)(ws + off); off += (size_t)HDIM * ND * sizeof(half_t);
  (void)in_sizes; (void)n_in; (void)out_size; (void)ws_size;

  dim3 blk(256);
  transpose_w_kernel<<<dim3(8, 8, 4), blk, 0, stream>>>(wq, wk, wv, wo, wTq, wTk, wTv, wTo);
  qkv_proj_kernel<<<dim3(BS / 128, ND / 64, 3), blk, 0, stream>>>(
      query, key, value, wTq, wTk, wTv, bq, bk, bv, phi_q, phi_k, v_h);
  chunk_kv_kernel<<<dim3(BN, NCHUNK), blk, 0, stream>>>(phi_k, v_h, kv, zc);
  scan_kernel<<<dim3(BN), blk, 0, stream>>>(kv, zc, stH, zx);
  attn_kernel<<<dim3(BN, NCHUNK), blk, 0, stream>>>(phi_q, phi_k, v_h, stH, zx, av);
  out_proj_kernel<<<dim3(BS / 128, HDIM / 64), blk, 0, stream>>>(av, wTo, bo, out);
}
